// FeatureLeaner_38594576122401
// MI455X (gfx1250) — matrix operand order repaired and checked by reading, not compiled, not run on hardware
//
#include <hip/hip_runtime.h>
#include <hip/hip_bf16.h>
#include <math.h>

typedef _Float16 hv16 __attribute__((ext_vector_type(16)));
typedef _Float16 hv8  __attribute__((ext_vector_type(8)));
typedef float    fv8  __attribute__((ext_vector_type(8)));

#define Bn   4
#define Cn   128
#define Hn   124
#define Wn   124
#define Ln   24
#define NPOS (Bn * Ln * Ln)
#define KTOT 1152            /* C * 9 reduction length               */
#define MH_STRIDE 1168       /* padded row stride in halves (32B-aligned rows) */
#define EPSF 1e-8f

// ---------------------------------------------------------------------------
// Kernel 1: per-position masked-cosine similarity via WMMA Gram matrix.
// One wave (32 lanes) per (b,i,j). Builds f16 matrix Mh[16][1152]:
//   rows 0..8 = masked condition patches (9 search shifts), row 9 = content,
//   rows 10..15 = 0.  D = Mh * Mh^T via 36x v_wmma_f32_16x16x32_f16.
// ---------------------------------------------------------------------------
__global__ __launch_bounds__(32)
void sim_argmax_kernel(const float* __restrict__ content,
                       const float* __restrict__ mask,
                       const float* __restrict__ condition,
                       float* __restrict__ maxval,
                       int*   __restrict__ rawidx)
{
    __shared__ __align__(32) _Float16 Mh[16][MH_STRIDE];
    __shared__ float Dm[16][16];

    const int lane = threadIdx.x;
    const int pos  = blockIdx.x;
    const int b    = pos / (Ln * Ln);
    const int rem  = pos - b * Ln * Ln;
    const int i    = rem / Ln;
    const int j    = rem - i * Ln;

    // Zero rows 10..15 (unused A rows / B columns must be 0 for the Gram).
    {
        unsigned int* z = (unsigned int*)&Mh[10][0];
        const int nd = 6 * MH_STRIDE / 2;
        for (int t = lane; t < nd; t += 32) z[t] = 0u;
    }

    // Build rows 0..9. Each lane handles channels lane, lane+32, lane+64, lane+96.
    for (int cc = 0; cc < Cn; cc += 32) {
        const int c    = cc + lane;
        const int base = ((b * Cn + c) * Hn + i * 5) * Wn + j * 5;
        float cont9[9], m9[9], cond25[25];
#pragma unroll
        for (int u = 0; u < 3; ++u)
#pragma unroll
            for (int v = 0; v < 3; ++v) {
                cont9[u * 3 + v] = content[base + (2 * u) * Wn + 2 * v];
                m9[u * 3 + v]    = mask   [base + (2 * u) * Wn + 2 * v];
            }
#pragma unroll
        for (int a = 0; a < 5; ++a)
#pragma unroll
            for (int e = 0; e < 5; ++e)
                cond25[a * 5 + e] = condition[base + (2 * a) * Wn + 2 * e];

        const int kc = c * 9;
#pragma unroll
        for (int r = 0; r < 9; ++r)
            Mh[9][kc + r] = (_Float16)cont9[r];
#pragma unroll
        for (int n = 0; n < 9; ++n) {
            const int sh = n / 3, sw = n - sh * 3;
#pragma unroll
            for (int u = 0; u < 3; ++u)
#pragma unroll
                for (int v = 0; v < 3; ++v) {
                    const int r = u * 3 + v;
                    const float val = (m9[r] > 0.f) ? cond25[(sh + u) * 5 + (sw + v)] : 0.f;
                    Mh[n][kc + r] = (_Float16)val;
                }
        }
    }
    __syncthreads();

    // Gram: D[m][n] = sum_k Mh[m][k] * Mh[n][k].
    // A fragment (16x32 f16): lane<16 -> row lane, K kb+0..7 / kb+16..23;
    //                         lane>=16 -> row lane-16, K kb+8..15 / kb+24..31.
    // B fragment (32x16 f16): lane holds column lane&15; same K order as A,
    //                         lanes<16 -> K kb+0..7 / kb+16..23, lanes>=16 -> K kb+8..15 / kb+24..31.
    const int mrow = lane & 15;
    const int aoff = (lane < 16) ? 0 : 8;
    const int boff = (lane < 16) ? 0 : 16;
    fv8 acc = {};
#pragma unroll 4
    for (int t = 0; t < KTOT / 32; ++t) {
        const int kb = t * 32;
        hv8 lo = *(const hv8*)&Mh[mrow][kb + aoff];
        hv8 hi = *(const hv8*)&Mh[mrow][kb + 16 + aoff];
        hv16 A;
#pragma unroll
        for (int e = 0; e < 8; ++e) { A[e] = lo[e]; A[8 + e] = hi[e]; }
        hv16 Bf = A;
        acc = __builtin_amdgcn_wmma_f32_16x16x32_f16(
                  /*neg_a=*/false, A, /*neg_b=*/false, Bf,
                  /*c_mod=*/(short)0, acc, /*reuse_a=*/false, /*reuse_b=*/false);
    }

    // Scatter accumulator (lane<16: M=r, lane>=16: M=r+8; N = lane&15).
    const int mbase = (lane < 16) ? 0 : 8;
#pragma unroll
    for (int r = 0; r < 8; ++r) Dm[mbase + r][mrow] = acc[r];
    __syncthreads();

    const float sq_a = Dm[9][9];
    const float na   = sqrtf(fmaxf(sq_a, 0.f)) + EPSF;
    float sim = -3.0e38f;
    if (lane < 9) {
        const float sq_b = Dm[lane][lane];
        const float dotv = Dm[9][lane];
        const float nb   = sqrtf(fmaxf(sq_b, 0.f)) + EPSF;
        float eud2 = sq_a / (na * na) + sq_b / (nb * nb) - 2.f * dotv / (na * nb);
        eud2 = fmaxf(eud2, 0.f);
        sim = (2.f - sqrtf(eud2)) * 0.5f;
    }
    // argmax with first-occurrence tie-break (matches jnp.argmax).
    float bestSim = sim;
    int   bestIdx = lane;
#pragma unroll
    for (int off = 16; off > 0; off >>= 1) {
        const float os = __shfl_xor(bestSim, off, 32);
        const int   oi = __shfl_xor(bestIdx, off, 32);
        if (os > bestSim || (os == bestSim && oi < bestIdx)) { bestSim = os; bestIdx = oi; }
    }
    if (lane == 0) {
        maxval[pos] = bestSim;
        rawidx[pos] = (bestSim > 0.f) ? bestIdx : 0;
    }
}

// ---------------------------------------------------------------------------
// Kernel 2: 4 slice means over overlapping column slabs of rawidx
//   piece p: columns [3p, 6p+6) for p<3, [9, 24) for p=3.
//   mean = rint(sum / (count(>0) + 1e-8))   (rint == jnp.round, RNE)
// ---------------------------------------------------------------------------
__global__ __launch_bounds__(32)
void slice_mean_kernel(const int* __restrict__ rawidx, float* __restrict__ sliceMean)
{
    const int lane = threadIdx.x;
#pragma unroll
    for (int p = 0; p < 4; ++p) {
        const int lo    = p * 3;
        const int hi    = (p < 3) ? (p * 6 + 6) : Ln;
        const int width = hi - lo;
        const int total = Bn * Ln * width;
        float s = 0.f, cnt = 0.f;
        for (int e = lane; e < total; e += 32) {
            const int jj = lo + (e % width);
            const int t  = e / width;
            const int ii = t % Ln;
            const int bb = t / Ln;
            const int v  = rawidx[(bb * Ln + ii) * Ln + jj];
            s += (float)v;
            if (v > 0) cnt += 1.f;
        }
#pragma unroll
        for (int off = 16; off > 0; off >>= 1) {
            s   += __shfl_xor(s,   off, 32);
            cnt += __shfl_xor(cnt, off, 32);
        }
        if (lane == 0) sliceMean[p] = rintf(s / (cnt + 1e-8f));
    }
}

// ---------------------------------------------------------------------------
// Kernel 3: fold is overlap-free (STRIDE==PATCH==5) -> pure gather/broadcast.
// Writes every element of both outputs (border rows/cols 120..123 are zero).
// ---------------------------------------------------------------------------
__global__ void finalize_kernel(const float* __restrict__ condition,
                                const float* __restrict__ maxval,
                                const int*   __restrict__ rawidx,
                                const float* __restrict__ sliceMean,
                                float* __restrict__ out)
{
    const int Ntot = Bn * Cn * Hn * Wn;
    const int idx = blockIdx.x * blockDim.x + threadIdx.x;
    if (idx >= Ntot) return;
    const int w = idx % Wn;
    int t = idx / Wn;
    const int h = t % Hn; t /= Hn;
    const int c = t % Cn;
    const int b = t / Cn;

    float mapped = 0.f, simv = 0.f;
    if (h < Ln * 5 && w < Ln * 5) {
        const int i = h / 5, p = h - i * 5;
        const int j = w / 5, q = w - j * 5;
        const int pos = (b * Ln + i) * Ln + j;
        int n = rawidx[pos];
        if (n == 0) n = (int)sliceMean[j / 6];   // zero index -> slice mean
        const int sh = n / 3, sw = n - sh * 3;
        mapped = condition[((b * Cn + c) * Hn + (i * 5 + sh * 2 + p)) * Wn
                           + (j * 5 + sw * 2 + q)];
        simv = maxval[pos];
    }
    out[idx]        = mapped;
    out[Ntot + idx] = simv;
}

// ---------------------------------------------------------------------------
extern "C" void kernel_launch(void* const* d_in, const int* in_sizes, int n_in,
                              void* d_out, int out_size, void* d_ws, size_t ws_size,
                              hipStream_t stream)
{
    (void)in_sizes; (void)n_in; (void)out_size; (void)ws_size;
    const float* content   = (const float*)d_in[0];
    const float* mask      = (const float*)d_in[1];
    const float* condition = (const float*)d_in[2];
    float* out = (float*)d_out;

    float* maxval    = (float*)d_ws;
    int*   rawidx    = (int*)((char*)d_ws + (size_t)NPOS * sizeof(float));
    float* sliceMean = (float*)((char*)d_ws + (size_t)2 * NPOS * sizeof(float));

    sim_argmax_kernel<<<NPOS, 32, 0, stream>>>(content, mask, condition, maxval, rawidx);
    slice_mean_kernel<<<1, 32, 0, stream>>>(rawidx, sliceMean);

    const int Ntot = Bn * Cn * Hn * Wn;
    finalize_kernel<<<(Ntot + 255) / 256, 256, 0, stream>>>(condition, maxval, rawidx,
                                                            sliceMean, out);
}
